// IterativeSE3Transformer_33517924778685
// MI455X (gfx1250) — compile-verified, matrix operand order checked by reading, not run on hardware
//
#include <hip/hip_runtime.h>
#include <hip/hip_fp16.h>

#define N_NODES 6144
#define C_CH    32
#define K_NBR   16
#define E_EDGES (N_NODES * K_NBR)   // 98304
#define EPSF    1e-6f

typedef __attribute__((ext_vector_type(2)))  float    v2f;
typedef __attribute__((ext_vector_type(8)))  float    v8f;
typedef __attribute__((ext_vector_type(16))) _Float16 v16h;

// ---------------------------------------------------------------- utilities
__device__ inline float wave_sum32(float v) {
#pragma unroll
  for (int o = 16; o > 0; o >>= 1) v += __shfl_xor(v, o, 32);
  return v;
}

// ------------------------------------------------------------ f32 GEMM tile
// Y[16 x ncols] = X[16 x 32] @ W[32 x ncols]; X row stride 32, W/Y stride ncols.
// Uses V_WMMA_F32_16X16X4_F32 (K=4 f32) when available, VALU fallback otherwise.
__device__ inline void gemm_tile_k32(const float* __restrict__ X,
                                     const float* __restrict__ W,
                                     float* __restrict__ Y, int ncols) {
  const int lane = threadIdx.x & 31;
  const int half = lane >> 4;
  const int l16  = lane & 15;
#if defined(__AMDGCN__) && __has_builtin(__builtin_amdgcn_wmma_f32_16x16x4_f32)
  for (int nt = 0; nt < (ncols >> 4); ++nt) {
    const int col = nt * 16 + l16;
    v8f acc = {};
#pragma unroll
    for (int kt = 0; kt < 8; ++kt) {
      const int k0 = kt * 4 + half * 2;
      v2f a, b;
      a.x = X[l16 * 32 + k0];
      a.y = X[l16 * 32 + k0 + 1];
      b.x = W[k0 * ncols + col];
      b.y = W[(k0 + 1) * ncols + col];
      acc = __builtin_amdgcn_wmma_f32_16x16x4_f32(false, a, false, b,
                                                  (short)0, acc, false, false);
    }
#pragma unroll
    for (int r = 0; r < 8; ++r)
      Y[(r + half * 8) * ncols + col] = acc[r];
  }
#else
  for (int nt = 0; nt < (ncols >> 4); ++nt) {
    const int col = nt * 16 + l16;
#pragma unroll
    for (int r = 0; r < 8; ++r) {
      const int row = r + half * 8;
      float s = 0.f;
      for (int k = 0; k < 32; ++k) s += X[row * 32 + k] * W[k * ncols + col];
      Y[row * ncols + col] = s;
    }
  }
#endif
}

// ------------------------------------------------------------------- init
__global__ void init_kernel(const float* __restrict__ pos_in,
                            const float* __restrict__ nf,
                            float* __restrict__ pos,
                            float* __restrict__ h0,
                            float* __restrict__ h1) {
  int t = blockIdx.x * blockDim.x + threadIdx.x;
  if (t < N_NODES * 3) pos[t] = pos_in[t];
  if (t < N_NODES * 32) h0[t] = nf[t];
  if (t < 3 * N_NODES * 32) h1[t] = 0.f;
}

// -------------------------------------------------------------------- KNN
// One thread per dst node; src positions tiled through LDS; register-guarded
// insertion into a private top-16 list (ascending distance).
__global__ void knn_kernel(const float* __restrict__ pos, int* __restrict__ nbr) {
  __shared__ float sp[256 * 3];
  const int dst = blockIdx.x * 256 + threadIdx.x;
  const float px = pos[dst * 3 + 0];
  const float py = pos[dst * 3 + 1];
  const float pz = pos[dst * 3 + 2];
  float bd[16];
  int   bi[16];
#pragma unroll
  for (int j = 0; j < 16; ++j) { bd[j] = 3.4e38f; bi[j] = 0; }
  float worst = 3.4e38f;
  for (int base = 0; base < N_NODES; base += 256) {
    __syncthreads();
    for (int i = threadIdx.x; i < 256 * 3; i += 256) sp[i] = pos[base * 3 + i];
    __syncthreads();
    for (int k = 0; k < 256; ++k) {
      const int s = base + k;
      const float dx = px - sp[k * 3 + 0];
      const float dy = py - sp[k * 3 + 1];
      const float dz = pz - sp[k * 3 + 2];
      const float d2 = dx * dx + dy * dy + dz * dz;
      if (s != dst && d2 < worst) {
        int p = 15;
        while (p > 0 && bd[p - 1] > d2) {
          bd[p] = bd[p - 1]; bi[p] = bi[p - 1]; --p;
        }
        bd[p] = d2; bi[p] = s;
        worst = bd[15];
      }
    }
  }
#pragma unroll
  for (int j = 0; j < 16; ++j) nbr[dst * 16 + j] = bi[j];
}

// ------------------------------------------------------------ edge geometry
__global__ void edge_geom_kernel(const float* __restrict__ pos,
                                 const int* __restrict__ nbr,
                                 float* __restrict__ dirv,
                                 float* __restrict__ ef) {
  const int e = blockIdx.x * blockDim.x + threadIdx.x;
  if (e >= E_EDGES) return;
  const int s = nbr[e];
  const int d = e >> 4;
  const float rx = pos[d * 3 + 0] - pos[s * 3 + 0];
  const float ry = pos[d * 3 + 1] - pos[s * 3 + 1];
  const float rz = pos[d * 3 + 2] - pos[s * 3 + 2];
  const float r  = sqrtf(rx * rx + ry * ry + rz * rz + EPSF);
  const float inv = 1.f / r;
  dirv[e * 3 + 0] = rx * inv;
  dirv[e * 3 + 1] = ry * inv;
  dirv[e * 3 + 2] = rz * inv;
#pragma unroll
  for (int i = 0; i < 16; ++i) {
    const float c = 4.f * (float)i / 15.f;
    const float t = r - c;
    ef[e * 17 + i] = __expf(-t * t * 2.f);   // exp(-(r-c)^2 / 0.5)
  }
  ef[e * 17 + 16] = r;
}

// -------------------------------------------------- radial MLP (f16 WMMA)
// rad[E][128] = relu(ef[E][17] @ W1[17][32] + b1) @ W2[32][128] + b2
// One wave per 16-edge tile; two staged f16 LDS tiles; K padded to 32.
__global__ void radial_kernel(const float* __restrict__ ef,
                              const float* __restrict__ W1,
                              const float* __restrict__ b1,
                              const float* __restrict__ W2,
                              const float* __restrict__ b2,
                              float* __restrict__ rad) {
  __shared__ _Float16 efh[4][16][32];
  __shared__ _Float16 hidh[4][16][32];
  const int wave = threadIdx.x >> 5;
  const int lane = threadIdx.x & 31;
  const int half = lane >> 4;
  const int l16  = lane & 15;
  const int tile = blockIdx.x * 4 + wave;
  const int e0   = tile * 16;

  for (int i = lane; i < 512; i += 32) {
    const int r = i >> 5, c = i & 31;
    const float v = (c < 17) ? ef[(size_t)(e0 + r) * 17 + c] : 0.f;
    efh[wave][r][c] = (_Float16)v;
  }
  v16h a;
#pragma unroll
  for (int j = 0; j < 16; ++j) {
    const int k = (j < 8) ? (half * 8 + j) : (16 + half * 8 + (j - 8));
    a[j] = efh[wave][l16][k];
  }
#pragma unroll
  for (int nt = 0; nt < 2; ++nt) {
    const int col = nt * 16 + l16;
    v16h b;
#pragma unroll
    for (int j = 0; j < 16; ++j) {
      const int k = (j < 8) ? (half * 8 + j) : (16 + half * 8 + (j - 8));
      b[j] = (k < 17) ? (_Float16)W1[k * 32 + col] : (_Float16)0.f;
    }
    v8f acc = {};
    acc = __builtin_amdgcn_wmma_f32_16x16x32_f16(false, a, false, b,
                                                 (short)0, acc, false, false);
#pragma unroll
    for (int r = 0; r < 8; ++r) {
      const int row = r + half * 8;
      hidh[wave][row][col] = (_Float16)fmaxf(acc[r] + b1[col], 0.f);
    }
  }
  v16h a2;
#pragma unroll
  for (int j = 0; j < 16; ++j) {
    const int k = (j < 8) ? (half * 8 + j) : (16 + half * 8 + (j - 8));
    a2[j] = hidh[wave][l16][k];
  }
#pragma unroll
  for (int nt = 0; nt < 8; ++nt) {
    const int col = nt * 16 + l16;
    v16h b;
#pragma unroll
    for (int j = 0; j < 16; ++j) {
      const int k = (j < 8) ? (half * 8 + j) : (16 + half * 8 + (j - 8));
      b[j] = (_Float16)W2[k * 128 + col];
    }
    v8f acc = {};
    acc = __builtin_amdgcn_wmma_f32_16x16x32_f16(false, a2, false, b,
                                                 (short)0, acc, false, false);
#pragma unroll
    for (int r = 0; r < 8; ++r) {
      const int row = r + half * 8;
      rad[(size_t)(e0 + row) * 128 + col] = acc[r] + b2[col];
    }
  }
}

// ---------------------------------------------- node projections (one layer)
// grid.y = 20 jobs: 5 deg-0 weights on h0, 5 deg-1 weights x 3 components.
// proj0: [q0,k0,v0,v01,s0][N][32];  proj1: [q1,k1,v1,v10,s1][3][N][32]
__global__ void proj_layer_kernel(const float* __restrict__ h0,
                                  const float* __restrict__ h1,
                                  const float* __restrict__ Wl,   // 10x32x32
                                  float* __restrict__ proj0,
                                  float* __restrict__ proj1) {
  const int wave = threadIdx.x >> 5;
  const int tile = blockIdx.x * (blockDim.x >> 5) + wave;   // 0..383
  const int job  = blockIdx.y;                              // 0..19
  const int NC   = N_NODES * 32;
  const int w0sel[5] = {0, 1, 2, 6, 8};   // Wq0 Wk0 Wv0 Wv01 Ws0
  const int w1sel[5] = {3, 4, 5, 7, 9};   // Wq1 Wk1 Wv1 Wv10 Ws1
  const float* X; const float* W; float* Y;
  if (job < 5) {
    W = Wl + w0sel[job] * 1024; X = h0; Y = proj0 + job * NC;
  } else {
    const int j = job - 5, slot = j / 3, comp = j % 3;
    W = Wl + w1sel[slot] * 1024;
    X = h1 + comp * NC;
    Y = proj1 + (slot * 3 + comp) * NC;
  }
  gemm_tile_k32(X + tile * 16 * 32, W, Y + tile * 16 * 32, 32);
}

// ---------------------------------------- attention over contiguous segments
// One wave per dst node, lane = channel. Softmax over its 16 edges is local.
__global__ void attn_edge_kernel(const int* __restrict__ nbr,
                                 const float* __restrict__ dirv,
                                 const float* __restrict__ rad,
                                 const float* __restrict__ proj0,
                                 const float* __restrict__ proj1,
                                 float* __restrict__ h0n,
                                 float* __restrict__ h1n) {
  const int n = blockIdx.x * (blockDim.x >> 5) + (threadIdx.x >> 5);
  const int c = threadIdx.x & 31;
  const int NC = N_NODES * 32;
  const float* q0p  = proj0;
  const float* k0p  = proj0 + 1 * NC;
  const float* v0p  = proj0 + 2 * NC;
  const float* v01p = proj0 + 3 * NC;
  const float* s0p  = proj0 + 4 * NC;
  const float* q1p  = proj1;
  const float* k1p  = proj1 + 3 * NC;
  const float* v1p  = proj1 + 6 * NC;
  const float* v10p = proj1 + 9 * NC;
  const float* s1p  = proj1 + 12 * NC;

  const float q0  = q0p[n * 32 + c];
  const float q1x = q1p[0 * NC + n * 32 + c];
  const float q1y = q1p[1 * NC + n * 32 + c];
  const float q1z = q1p[2 * NC + n * 32 + c];

  float lg[16];
#pragma unroll
  for (int j = 0; j < 16; ++j) {
    const int e = n * 16 + j;
    const int s = nbr[e];
    const float* re = rad + (size_t)e * 128;
    const float ak0 = re[c], ak1 = re[64 + c];
    float t = q0 * k0p[s * 32 + c] * ak0 +
              (q1x * k1p[0 * NC + s * 32 + c] +
               q1y * k1p[1 * NC + s * 32 + c] +
               q1z * k1p[2 * NC + s * 32 + c]) * ak1;
#pragma unroll
    for (int o = 16; o > 0; o >>= 1) t += __shfl_xor(t, o, 32);
    lg[j] = t * 0.125f;                       // / sqrt(2*C) = 8
  }
  float m = lg[0];
#pragma unroll
  for (int j = 1; j < 16; ++j) m = fmaxf(m, lg[j]);
  float z = 0.f;
#pragma unroll
  for (int j = 0; j < 16; ++j) { lg[j] = __expf(lg[j] - m); z += lg[j]; }
  const float zi = 1.f / (z + EPSF);

  float o0  = s0p[n * 32 + c];
  float o1x = s1p[0 * NC + n * 32 + c];
  float o1y = s1p[1 * NC + n * 32 + c];
  float o1z = s1p[2 * NC + n * 32 + c];
#pragma unroll
  for (int j = 0; j < 16; ++j) {
    const int e = n * 16 + j;
    const int s = nbr[e];
    const float* re = rad + (size_t)e * 128;
    const float av0 = re[32 + c], av1 = re[96 + c];
    const float w = lg[j] * zi;
    const float dx = dirv[e * 3 + 0], dy = dirv[e * 3 + 1], dz = dirv[e * 3 + 2];
    const float v10 = v10p[0 * NC + s * 32 + c] * dx +
                      v10p[1 * NC + s * 32 + c] * dy +
                      v10p[2 * NC + s * 32 + c] * dz;
    o0 += w * (v0p[s * 32 + c] * av0 + v10);
    const float v01 = v01p[s * 32 + c] * av0;
    o1x += w * (v1p[0 * NC + s * 32 + c] * av1 + v01 * dx);
    o1y += w * (v1p[1 * NC + s * 32 + c] * av1 + v01 * dy);
    o1z += w * (v1p[2 * NC + s * 32 + c] * av1 + v01 * dz);
  }
  h0n[n * 32 + c] = o0;
  h1n[0 * NC + n * 32 + c] = o1x;
  h1n[1 * NC + n * 32 + c] = o1y;
  h1n[2 * NC + n * 32 + c] = o1z;
}

// -------------------------------------------------------------- NormSE3
__global__ void norm_kernel(float* __restrict__ h0, float* __restrict__ h1,
                            const float* __restrict__ g, const float* __restrict__ b) {
  const int n = blockIdx.x * (blockDim.x >> 5) + (threadIdx.x >> 5);
  const int c = threadIdx.x & 31;
  const int NC = N_NODES * 32;
  const float x0 = h0[n * 32 + c];
  const float n0 = fabsf(x0) + EPSF;
  float mean = wave_sum32(n0) * (1.f / 32.f);
  float d    = n0 - mean;
  float var  = wave_sum32(d * d) * (1.f / 32.f);
  float ln   = d * rsqrtf(var + 1e-5f);
  const float s0 = fmaxf(ln * g[c] + b[c], 0.f);
  h0[n * 32 + c] = x0 / n0 * s0;

  const float hx = h1[0 * NC + n * 32 + c];
  const float hy = h1[1 * NC + n * 32 + c];
  const float hz = h1[2 * NC + n * 32 + c];
  const float n1 = sqrtf(hx * hx + hy * hy + hz * hz + EPSF);
  mean = wave_sum32(n1) * (1.f / 32.f);
  d    = n1 - mean;
  var  = wave_sum32(d * d) * (1.f / 32.f);
  ln   = d * rsqrtf(var + 1e-5f);
  const float s1 = fmaxf(ln * g[32 + c] + b[32 + c], 0.f);
  const float sc = s1 / n1;
  h1[0 * NC + n * 32 + c] = hx * sc;
  h1[1 * NC + n * 32 + c] = hy * sc;
  h1[2 * NC + n * 32 + c] = hz * sc;
}

// ------------------------------------------------------------- pos update
__global__ void pos_update_kernel(float* __restrict__ pos, const float* __restrict__ h1) {
  const int n = blockIdx.x * blockDim.x + threadIdx.x;
  if (n >= N_NODES) return;
  const int NC = N_NODES * 32;
  pos[n * 3 + 0] += h1[0 * NC + n * 32 + 0];
  pos[n * 3 + 1] += h1[1 * NC + n * 32 + 0];
  pos[n * 3 + 2] += h1[2 * NC + n * 32 + 0];
}

// ------------------------------------------------------ final projections
__global__ void final_proj_kernel(const float* __restrict__ h0,
                                  const float* __restrict__ h1,
                                  const float* __restrict__ fW00,
                                  const float* __restrict__ fW10,
                                  const float* __restrict__ fWs,
                                  float* __restrict__ A0, float* __restrict__ A1,
                                  float* __restrict__ S) {
  const int wave = threadIdx.x >> 5;
  const int tile = blockIdx.x * (blockDim.x >> 5) + wave;
  const int job  = blockIdx.y;            // 0..4
  const int NC   = N_NODES * 32;
  const float* X; const float* W; float* Y;
  if (job == 0)      { X = h0;                 W = fW00; Y = A0; }
  else if (job <= 3) { X = h1 + (job - 1) * NC; W = fW10; Y = A1 + (job - 1) * N_NODES * 64; }
  else               { X = h0;                 W = fWs;  Y = S; }
  gemm_tile_k32(X + tile * 16 * 32, W, Y + tile * 16 * 64, 64);
}

// ------------------------------------------------------ final edge reduce
__global__ void final_edge_kernel(const int* __restrict__ nbr,
                                  const float* __restrict__ dirv,
                                  const float* __restrict__ frad,
                                  const float* __restrict__ A0,
                                  const float* __restrict__ A1,
                                  const float* __restrict__ S,
                                  float* __restrict__ out) {
  const int t = blockIdx.x * blockDim.x + threadIdx.x;
  const int n = t >> 6, f = t & 63;
  if (n >= N_NODES) return;
  const int NF = N_NODES * 64;
  float acc = S[n * 64 + f];
#pragma unroll
  for (int j = 0; j < 16; ++j) {
    const int e = n * 16 + j;
    const int s = nbr[e];
    const float a0 = frad[(size_t)e * 128 + f];
    const float a1 = frad[(size_t)e * 128 + 64 + f];
    const float dx = dirv[e * 3 + 0], dy = dirv[e * 3 + 1], dz = dirv[e * 3 + 2];
    acc += A0[s * 64 + f] * a0 +
           (A1[0 * NF + s * 64 + f] * dx +
            A1[1 * NF + s * 64 + f] * dy +
            A1[2 * NF + s * 64 + f] * dz) * a1;
  }
  out[n * 64 + f] = acc;
}

// ---------------------------------------------------------------- launcher
extern "C" void kernel_launch(void* const* d_in, const int* in_sizes, int n_in,
                              void* d_out, int out_size, void* d_ws, size_t ws_size,
                              hipStream_t stream) {
  (void)in_sizes; (void)n_in; (void)out_size; (void)ws_size;
  const float* pos_in = (const float*)d_in[0];
  const float* nf     = (const float*)d_in[1];
  const float* attn_W = (const float*)d_in[2];
  const float* rad_W1 = (const float*)d_in[3];
  const float* rad_b1 = (const float*)d_in[4];
  const float* rad_W2 = (const float*)d_in[5];
  const float* rad_b2 = (const float*)d_in[6];
  const float* norm_g = (const float*)d_in[7];
  const float* norm_b = (const float*)d_in[8];
  const float* fW00   = (const float*)d_in[9];
  const float* fW10   = (const float*)d_in[10];
  const float* fWs    = (const float*)d_in[11];
  const float* frW1   = (const float*)d_in[12];
  const float* frb1   = (const float*)d_in[13];
  const float* frW2   = (const float*)d_in[14];
  const float* frb2   = (const float*)d_in[15];
  float* out = (float*)d_out;

  char* wp = (char*)d_ws;
  auto alloc = [&](size_t nbytes) -> void* {
    void* p = (void*)wp;
    wp += (nbytes + 255) & ~(size_t)255;
    return p;
  };
  const int NC = N_NODES * 32;
  float* pos   = (float*)alloc((size_t)N_NODES * 3 * 4);
  int*   nbr   = (int*)  alloc((size_t)E_EDGES * 4);
  float* dirv  = (float*)alloc((size_t)E_EDGES * 3 * 4);
  float* ef    = (float*)alloc((size_t)E_EDGES * 17 * 4);
  float* rad   = (float*)alloc((size_t)E_EDGES * 128 * 4);
  float* h0a   = (float*)alloc((size_t)NC * 4);
  float* h0b   = (float*)alloc((size_t)NC * 4);
  float* h1a   = (float*)alloc((size_t)3 * NC * 4);
  float* h1b   = (float*)alloc((size_t)3 * NC * 4);
  float* proj0 = (float*)alloc((size_t)5 * NC * 4);
  float* proj1 = (float*)alloc((size_t)15 * NC * 4);
  float* A0    = (float*)alloc((size_t)N_NODES * 64 * 4);
  float* A1    = (float*)alloc((size_t)3 * N_NODES * 64 * 4);
  float* S     = (float*)alloc((size_t)N_NODES * 64 * 4);

  float* h0 = h0a; float* h0n = h0b;
  float* h1 = h1a; float* h1n = h1b;

  init_kernel<<<(3 * NC + 255) / 256, 256, 0, stream>>>(pos_in, nf, pos, h0, h1);

  for (int it = 0; it < 2; ++it) {
    knn_kernel<<<N_NODES / 256, 256, 0, stream>>>(pos, nbr);
    edge_geom_kernel<<<E_EDGES / 256, 256, 0, stream>>>(pos, nbr, dirv, ef);
    for (int l = 0; l < 2; ++l) {
      const int li = it * 2 + l;
      radial_kernel<<<E_EDGES / 64, 128, 0, stream>>>(
          ef, rad_W1 + li * 17 * 32, rad_b1 + li * 32,
          rad_W2 + li * 32 * 128, rad_b2 + li * 128, rad);
      proj_layer_kernel<<<dim3(N_NODES / 16 / 8, 20), 256, 0, stream>>>(
          h0, h1, attn_W + (size_t)li * 10 * 1024, proj0, proj1);
      attn_edge_kernel<<<N_NODES / 8, 256, 0, stream>>>(
          nbr, dirv, rad, proj0, proj1, h0n, h1n);
      norm_kernel<<<N_NODES / 8, 256, 0, stream>>>(
          h0n, h1n, norm_g + li * 64, norm_b + li * 64);
      float* t;
      t = h0; h0 = h0n; h0n = t;
      t = h1; h1 = h1n; h1n = t;
    }
    pos_update_kernel<<<N_NODES / 256, 256, 0, stream>>>(pos, h1);
  }

  knn_kernel<<<N_NODES / 256, 256, 0, stream>>>(pos, nbr);
  edge_geom_kernel<<<E_EDGES / 256, 256, 0, stream>>>(pos, nbr, dirv, ef);
  radial_kernel<<<E_EDGES / 64, 128, 0, stream>>>(ef, frW1, frb1, frW2, frb2, rad);
  final_proj_kernel<<<dim3(N_NODES / 16 / 8, 5), 256, 0, stream>>>(
      h0, h1, fW00, fW10, fWs, A0, A1, S);
  final_edge_kernel<<<N_NODES * 64 / 256, 256, 0, stream>>>(
      nbr, dirv, rad, A0, A1, S, out);
}
